// GPT_65609920414255
// MI455X (gfx1250) — matrix operand order repaired and checked by reading, not compiled, not run on hardware
//
#include <hip/hip_runtime.h>
#include <hip/hip_bf16.h>
#include <stdint.h>

// GPT-2 small forward for gfx1250 (MI455X): all GEMMs + attention on
// v_wmma_f32_16x16x32_bf16 (wave32). fp32 weights streamed once from HBM,
// converted to bf16 in-register during LDS staging; activations bf16 (L2-
// resident); residual stream fp32. A-tile staging uses the CDNA5 async
// global->LDS path; W stream is prefetched one K-step ahead.

#define LCOUNT 12
#define HEADS  12
#define CDIM   768
#define DDIM   64
#define TDIM   1024
#define BDIM   2
#define MTOK   (BDIM*TDIM)   // 2048
#define VDIM   50257

typedef __attribute__((ext_vector_type(16))) __bf16 v16bf;
typedef __attribute__((ext_vector_type(8)))  float  v8f;
typedef __attribute__((ext_vector_type(8)))  int    v8i;

__device__ __forceinline__ uint16_t f2bf(float f){
  uint32_t u = __float_as_uint(f);
  u += 0x7FFFu + ((u >> 16) & 1u);       // round-to-nearest-even
  return (uint16_t)(u >> 16);
}
__device__ __forceinline__ uint4 ld4(const uint16_t* p){
  return *reinterpret_cast<const uint4*>(p);
}
__device__ __forceinline__ v16bf mkfrag(uint4 a, uint4 b){
  v8i t;
  t[0]=(int)a.x; t[1]=(int)a.y; t[2]=(int)a.z; t[3]=(int)a.w;
  t[4]=(int)b.x; t[5]=(int)b.y; t[6]=(int)b.z; t[7]=(int)b.w;
  return __builtin_bit_cast(v16bf, t);
}
__device__ __forceinline__ v8f wmma_bf16(v16bf a, v16bf b, v8f c){
  return __builtin_amdgcn_wmma_f32_16x16x32_bf16(false, a, false, b, (short)0, c, false, false);
}
__device__ __forceinline__ v8f vzero(){
  v8f z = {0.f,0.f,0.f,0.f,0.f,0.f,0.f,0.f};
  return z;
}
// CDNA5 async copy: global -> LDS, 16 bytes per lane (ASYNCcnt).
__device__ __forceinline__ void async_copy_b128(uint32_t lds_addr, const void* gaddr){
  asm volatile("global_load_async_to_lds_b128 %0, %1, off"
               :: "v"(lds_addr), "v"(gaddr) : "memory");
}
__device__ __forceinline__ void wait_asynccnt0(){
  asm volatile("s_wait_asynccnt 0x0" ::: "memory");
}

// ---------------- embedding: x = wte[idx] + wpe[t]  (fp32) ----------------
__global__ void embed_kernel(const int* __restrict__ idx, const float* __restrict__ wte,
                             const float* __restrict__ wpe, float* __restrict__ x){
  int row = blockIdx.x;             // token 0..2047
  int t = row % TDIM;
  int id = idx[row];
  const float* wt = wte + (size_t)id * CDIM;
  const float* wp = wpe + (size_t)t  * CDIM;
  float* xr = x + (size_t)row * CDIM;
  for (int c = threadIdx.x; c < CDIM; c += blockDim.x)
    xr[c] = wt[c] + wp[c];
}

// ---------------- layernorm (fp32 in -> bf16 out) ----------------
__global__ void ln_bf16_kernel(const float* __restrict__ x, const float* __restrict__ g,
                               const float* __restrict__ b, uint16_t* __restrict__ h){
  __shared__ float s1[256];
  __shared__ float s2[256];
  int row = blockIdx.x, tid = threadIdx.x;
  const float* xr = x + (size_t)row * CDIM;
  float v0 = xr[tid], v1 = xr[tid+256], v2 = xr[tid+512];
  s1[tid] = v0+v1+v2;
  s2[tid] = v0*v0 + v1*v1 + v2*v2;
  __syncthreads();
  for (int s = 128; s > 0; s >>= 1){
    if (tid < s){ s1[tid] += s1[tid+s]; s2[tid] += s2[tid+s]; }
    __syncthreads();
  }
  float mean = s1[0] * (1.0f/CDIM);
  float var  = s2[0] * (1.0f/CDIM) - mean*mean;
  float rs   = rsqrtf(var + 1e-5f);
  uint16_t* hr = h + (size_t)row * CDIM;
  hr[tid]     = f2bf((v0-mean)*rs*g[tid]     + b[tid]);
  hr[tid+256] = f2bf((v1-mean)*rs*g[tid+256] + b[tid+256]);
  hr[tid+512] = f2bf((v2-mean)*rs*g[tid+512] + b[tid+512]);
}

// ---------------- tiled WMMA GEMM: out = A(bf16)[2048,K] @ W(f32)[K,N] + bias ----------------
#define EPI_BF16  0   // store bf16
#define EPI_RESID 1   // outf = resid + acc + bias  (fp32)
#define EPI_GELU  2   // store bf16(gelu(acc+bias))
#define EPI_F32   3   // store fp32 (lm_head)
#define EPI_QKV   4   // store bf16; V columns also scattered transposed into vt

__device__ __forceinline__ float gelu_tanh(float x){
  float x3 = x*x*x;
  return 0.5f*x*(1.0f + tanhf(0.7978845608f*(x + 0.044715f*x3)));
}

template<int EPI>
__launch_bounds__(256)
__global__ void gemm_kernel(const uint16_t* __restrict__ A, const float* __restrict__ W,
                            const float* __restrict__ bias, const float* __restrict__ resid,
                            float* __restrict__ outf, uint16_t* __restrict__ outb,
                            uint16_t* __restrict__ vt,
                            int N, int K){
  // block tile 64(M) x 128(N), K-step 32; 8 waves: 4(M) x 2(N), each 16x64
  __shared__ uint16_t sA[64*40];    // [row][k], pad 8
  __shared__ uint16_t sW[128*40];   // [col][k] (transposed), pad 8
  int tid  = threadIdx.x;
  int lane = tid & 31, wid = tid >> 5;
  int lg   = lane >> 4, l16 = lane & 15;
  int wm   = wid >> 1,  wn  = wid & 1;
  int m0   = blockIdx.x * 64;
  int n0   = blockIdx.y * 128;

  v8f acc[4] = {vzero(), vzero(), vzero(), vzero()};

  int arow  = tid >> 2;             // 0..63
  int acol  = (tid & 3) * 8;        // 0,8,16,24
  int wcol  = tid & 127;            // 0..127
  int wrow0 = (tid >> 7) * 16;      // 0 or 16
  int colg  = n0 + wcol;
  uint32_t sa_dst = (uint32_t)(uintptr_t)&sA[arow*40 + acol];

  for (int k0 = 0; k0 < K; k0 += 32){
    // stage A (already bf16): async global->LDS, one b128 per thread
    async_copy_b128(sa_dst, A + (size_t)(m0 + arow)*K + k0 + acol);
    // prefetch next K-step of the fp32 weight stream
    if (k0 + 32 < K && colg < N)
      __builtin_prefetch(W + (size_t)(k0 + 32 + wrow0)*N + colg, 0, 1);
    // stage W transposed, fp32 -> bf16 in-register
    {
      uint32_t pk[8];
      #pragma unroll
      for (int i = 0; i < 8; ++i){
        float f0 = 0.f, f1 = 0.f;
        if (colg < N){
          f0 = W[(size_t)(k0 + wrow0 + 2*i    )*N + colg];
          f1 = W[(size_t)(k0 + wrow0 + 2*i + 1)*N + colg];
        }
        pk[i] = (uint32_t)f2bf(f0) | ((uint32_t)f2bf(f1) << 16);
      }
      *reinterpret_cast<uint4*>(&sW[wcol*40 + wrow0    ]) = make_uint4(pk[0],pk[1],pk[2],pk[3]);
      *reinterpret_cast<uint4*>(&sW[wcol*40 + wrow0 + 8]) = make_uint4(pk[4],pk[5],pk[6],pk[7]);
    }
    wait_asynccnt0();               // own async A copy complete
    __syncthreads();                // all waves' staging visible
    // compute: A frag (16x32 bf16 layout), 4 B frags, 4 WMMAs
    int kb = lg*8;
    const uint16_t* ap = &sA[(wm*16 + l16)*40];
    v16bf af = mkfrag(ld4(ap + kb), ld4(ap + 16 + kb));
    #pragma unroll
    for (int nt = 0; nt < 4; ++nt){
      const uint16_t* bp = &sW[(wn*64 + nt*16 + l16)*40 + lg*8];
      v16bf bf = mkfrag(ld4(bp), ld4(bp + 16));
      acc[nt] = wmma_bf16(af, bf, acc[nt]);
    }
    __syncthreads();
  }

  // epilogue (C-layout: row = j + 8*lg, col = l16)
  #pragma unroll
  for (int nt = 0; nt < 4; ++nt){
    int cg = n0 + wn*64 + nt*16 + l16;
    if (cg >= N) continue;
    float bv = (bias != nullptr) ? bias[cg] : 0.f;
    #pragma unroll
    for (int j = 0; j < 8; ++j){
      int rg = m0 + wm*16 + j + 8*lg;
      float v = acc[nt][j] + bv;
      size_t o = (size_t)rg * N + cg;
      if (EPI == EPI_BF16){
        outb[o] = f2bf(v);
      } else if (EPI == EPI_QKV){
        uint16_t bvv = f2bf(v);
        outb[o] = bvv;
        if (cg >= 2*CDIM){          // V part: also store transposed vt[b][c][t]
          int bb = rg >> 10, tt = rg & (TDIM-1);
          vt[((size_t)bb*CDIM + (cg - 2*CDIM))*TDIM + tt] = bvv;
        }
      } else if (EPI == EPI_GELU){
        outb[o] = f2bf(gelu_tanh(v));
      } else if (EPI == EPI_RESID){
        outf[o] = resid[o] + v;
      } else {
        outf[o] = v;
      }
    }
  }
}

// ---------------- flash attention (causal, bf16 WMMA, online softmax) ----------------
__launch_bounds__(128)
__global__ void attn_kernel(const uint16_t* __restrict__ qkv, const uint16_t* __restrict__ vt,
                            uint16_t* __restrict__ y){
  __shared__ uint16_t lds_p[4][16][40];   // per-wave P tile (16x32 bf16)
  int tid  = threadIdx.x;
  int lane = tid & 31, wid = tid >> 5;
  int lg   = lane >> 4, l16 = lane & 15;
  int bid  = blockIdx.x;
  int qb   = bid & 15;
  int hh   = (bid >> 4) % HEADS;
  int bb   = bid / (16*HEADS);
  int tokbase = bb * TDIM;
  int q0   = qb*64 + wid*16;              // wave's 16 q-rows
  const float scale = 0.125f;             // 1/sqrt(64)
  const int QS = 3*CDIM;                  // 2304 row stride in qkv

  // Q fragments (A layout), d 0..31 and 32..63
  const uint16_t* qp = qkv + (size_t)(tokbase + q0 + l16)*QS + hh*DDIM;
  int kb = lg*8;
  v16bf q_lo = mkfrag(ld4(qp + kb),      ld4(qp + 16 + kb));
  v16bf q_hi = mkfrag(ld4(qp + 32 + kb), ld4(qp + 48 + kb));

  float m_i[8], l_i[8];
  v8f o_acc[4] = {vzero(), vzero(), vzero(), vzero()};
  #pragma unroll
  for (int j = 0; j < 8; ++j){ m_i[j] = -3.0e38f; l_i[j] = 0.f; }

  int nch = (q0 + 16 + 31) >> 5;          // 32-key chunks (causal bound)
  for (int jc = 0; jc < nch; ++jc){
    int j0 = jc << 5;
    v8f s[2];
    #pragma unroll
    for (int nt = 0; nt < 2; ++nt){
      // B = K^T: lane's d-values (8*lg..+7, 16+8*lg..+7) of key (j0+nt*16+l16)
      const uint16_t* kp = qkv + (size_t)(tokbase + j0 + nt*16 + l16)*QS + CDIM + hh*DDIM + lg*8;
      v16bf k_lo = mkfrag(ld4(kp),      ld4(kp + 16));
      v16bf k_hi = mkfrag(ld4(kp + 32), ld4(kp + 48));
      v8f sv = vzero();
      sv = wmma_bf16(q_lo, k_lo, sv);
      sv = wmma_bf16(q_hi, k_hi, sv);
      s[nt] = sv;
    }
    // prefetch next chunk's K rows
    if (j0 + 32 < q0 + 16)
      __builtin_prefetch(qkv + (size_t)(tokbase + j0 + 32 + l16)*QS + CDIM + hh*DDIM, 0, 1);
    // causal mask + online softmax (row = j + 8*lg, cols across 16-lane group)
    #pragma unroll
    for (int j = 0; j < 8; ++j){
      int qrow = q0 + j + 8*lg;
      float s0 = s[0][j]*scale, s1 = s[1][j]*scale;
      if (j0 + l16 > qrow)      s0 = -3.0e38f;
      if (j0 + 16 + l16 > qrow) s1 = -3.0e38f;
      float cm = fmaxf(s0, s1);
      #pragma unroll
      for (int mm = 1; mm < 16; mm <<= 1) cm = fmaxf(cm, __shfl_xor(cm, mm, 32));
      float mnew = fmaxf(m_i[j], cm);
      float p0 = __expf(s0 - mnew), p1 = __expf(s1 - mnew);
      float rs = p0 + p1;
      #pragma unroll
      for (int mm = 1; mm < 16; mm <<= 1) rs += __shfl_xor(rs, mm, 32);
      float sf = __expf(m_i[j] - mnew);
      l_i[j] = l_i[j]*sf + rs;
      m_i[j] = mnew;
      #pragma unroll
      for (int nt2 = 0; nt2 < 4; ++nt2) o_acc[nt2][j] *= sf;
      lds_p[wid][j + 8*lg][l16]      = f2bf(p0);
      lds_p[wid][j + 8*lg][16 + l16] = f2bf(p1);
    }
    // P (C-layout in LDS) -> A fragment
    const uint16_t* pp = &lds_p[wid][l16][0];
    v16bf pa = mkfrag(ld4(pp + kb), ld4(pp + 16 + kb));
    // O += P @ V : V read key-major from transposed vt -> 2 b128 per fragment
    #pragma unroll
    for (int nt2 = 0; nt2 < 4; ++nt2){
      const uint16_t* vrow = vt + ((size_t)bb*CDIM + hh*DDIM + nt2*16 + l16)*TDIM + j0 + lg*8;
      v16bf vb = mkfrag(ld4(vrow), ld4(vrow + 16));
      o_acc[nt2] = wmma_bf16(pa, vb, o_acc[nt2]);
    }
  }
  // normalize + write y (bf16)
  #pragma unroll
  for (int j = 0; j < 8; ++j){
    float inv = 1.0f / l_i[j];
    int tok = tokbase + q0 + j + 8*lg;
    #pragma unroll
    for (int nt2 = 0; nt2 < 4; ++nt2){
      y[(size_t)tok*CDIM + hh*DDIM + nt2*16 + l16] = f2bf(o_acc[nt2][j] * inv);
    }
  }
}

// ---------------- host orchestration ----------------
extern "C" void kernel_launch(void* const* d_in, const int* in_sizes, int n_in,
                              void* d_out, int out_size, void* d_ws, size_t ws_size,
                              hipStream_t stream){
  (void)in_sizes; (void)n_in; (void)out_size; (void)ws_size;
  const int*   idx   = (const int*)  d_in[0];
  const float* wte   = (const float*)d_in[1];
  const float* wpe   = (const float*)d_in[2];
  const float* ln1_g = (const float*)d_in[3];
  const float* ln1_b = (const float*)d_in[4];
  const float* wqkv  = (const float*)d_in[5];
  const float* bqkv  = (const float*)d_in[6];
  const float* wproj = (const float*)d_in[7];
  const float* bproj = (const float*)d_in[8];
  const float* ln2_g = (const float*)d_in[9];
  const float* ln2_b = (const float*)d_in[10];
  const float* wfc   = (const float*)d_in[11];
  const float* bfc   = (const float*)d_in[12];
  const float* wfc2  = (const float*)d_in[13];
  const float* bfc2  = (const float*)d_in[14];
  const float* lnf_g = (const float*)d_in[15];
  const float* lnf_b = (const float*)d_in[16];
  const float* lm_w  = (const float*)d_in[17];
  float* out = (float*)d_out;

  // workspace layout (bytes): ~37.7 MB total
  uint8_t* ws = (uint8_t*)d_ws;
  float*    x    = (float*)   (ws);                       // 2048x768 f32   = 6291456
  uint16_t* h    = (uint16_t*)(ws +  6291456);            // 2048x768 bf16  = 3145728
  uint16_t* qkvb = (uint16_t*)(ws +  9437184);            // 2048x2304 bf16 = 9437184
  uint16_t* yb   = (uint16_t*)(ws + 18874368);            // 2048x768 bf16  = 3145728
  uint16_t* gb   = (uint16_t*)(ws + 22020096);            // 2048x3072 bf16 = 12582912
  uint16_t* vt   = (uint16_t*)(ws + 34603008);            // 2x768x1024 bf16= 3145728

  embed_kernel<<<MTOK, 256, 0, stream>>>(idx, wte, wpe, x);
  for (int l = 0; l < LCOUNT; ++l){
    ln_bf16_kernel<<<MTOK, 256, 0, stream>>>(x, ln1_g + l*CDIM, ln1_b + l*CDIM, h);
    gemm_kernel<EPI_QKV><<<dim3(32, 18), 256, 0, stream>>>(
        h, wqkv + (size_t)l*CDIM*3*CDIM, bqkv + (size_t)l*3*CDIM,
        nullptr, nullptr, qkvb, vt, 3*CDIM, CDIM);
    attn_kernel<<<BDIM*HEADS*16, 128, 0, stream>>>(qkvb, vt, yb);
    gemm_kernel<EPI_RESID><<<dim3(32, 6), 256, 0, stream>>>(
        yb, wproj + (size_t)l*CDIM*CDIM, bproj + (size_t)l*CDIM,
        x, x, nullptr, nullptr, CDIM, CDIM);
    ln_bf16_kernel<<<MTOK, 256, 0, stream>>>(x, ln2_g + l*CDIM, ln2_b + l*CDIM, h);
    gemm_kernel<EPI_GELU><<<dim3(32, 24), 256, 0, stream>>>(
        h, wfc + (size_t)l*CDIM*4*CDIM, bfc + (size_t)l*4*CDIM,
        nullptr, nullptr, gb, nullptr, 4*CDIM, CDIM);
    gemm_kernel<EPI_RESID><<<dim3(32, 6), 256, 0, stream>>>(
        gb, wfc2 + (size_t)l*4*CDIM*CDIM, bfc2 + (size_t)l*CDIM,
        x, x, nullptr, nullptr, CDIM, 4*CDIM);
  }
  ln_bf16_kernel<<<MTOK, 256, 0, stream>>>(x, lnf_g, lnf_b, h);
  gemm_kernel<EPI_F32><<<dim3(32, (VDIM + 127)/128), 256, 0, stream>>>(
      h, lm_w, nullptr, nullptr, out, nullptr, nullptr, VDIM, CDIM);
}
